// GaussianWindowAttention_44315472560839
// MI455X (gfx1250) — matrix operand order repaired and checked by reading, not compiled, not run on hardware
//
#include <hip/hip_runtime.h>
#include <hip/hip_bf16.h>

// ---------------------------------------------------------------------------
// GaussianWindowAttention on MI455X (gfx1250), wave32 + WMMA f16->f32.
// B=2048, N=64 tokens (8x8 window), DIM=512, HEADS=16, HEAD_DIM=32.
// ---------------------------------------------------------------------------

typedef _Float16 f16;
typedef __attribute__((ext_vector_type(16))) _Float16 v16h;
typedef __attribute__((ext_vector_type(8)))  _Float16 v8h;
typedef __attribute__((ext_vector_type(4)))  _Float16 v4h;
typedef __attribute__((ext_vector_type(8)))  float    v8f;
typedef __attribute__((ext_vector_type(4)))  int      v4i;

union V16U { v16h v; v8h h[2]; };

#define DIMC     512
#define NTOK     64
#define HEADS    16
#define HEADDIM  32
#define BATCH    2048

// A-fragment (16x32 f16, M x K) from a row-major tile, leading dim ld (f16).
static __device__ inline v16h load_frag_a(const f16* base, int lane, int ld) {
  const int half = lane >> 4;
  const int r    = lane & 15;
  const f16* p   = base + r * ld;
  V16U u;
  u.h[0] = *(const v8h*)(p + half * 8);
  u.h[1] = *(const v8h*)(p + 16 + half * 8);
  return u.v;
}

// B-fragment (32x16 f16, K x N) supplied as B^T row-major [N][K], ld in f16.
static __device__ inline v16h load_frag_b(const f16* bt, int lane, int ld) {
  const int half = lane >> 4;
  const int n    = lane & 15;
  const f16* p   = bt + n * ld + half * 8;
  V16U u;
  u.h[0] = *(const v8h*)(p);
  u.h[1] = *(const v8h*)(p + 16);
  return u.v;
}

static __device__ inline v8f wmma_f16(v16h a, v16h b, v8f c) {
  return __builtin_amdgcn_wmma_f32_16x16x32_f16(false, a, false, b,
                                                (short)0, c, false, false);
}

static __device__ inline float dist2(int i, int j) {
  const int dy = (i >> 3) - (j >> 3);
  const int dx = (i & 7) - (j & 7);
  return (float)(dy * dy + dx * dx);
}

// ---------------------------------------------------------------------------
// Kernel 0: convert weights fp32 -> f16, transposed to [N][K] for B-fragments.
// ---------------------------------------------------------------------------
__global__ void gwa_prep_weights(const float* __restrict__ wqkv,
                                 const float* __restrict__ wproj,
                                 f16* __restrict__ wqkvT,
                                 f16* __restrict__ wprojT) {
  int idx = blockIdx.x * 256 + threadIdx.x;
  const int NQ = 3 * DIMC * DIMC;   // 1536*512
  if (idx < NQ) {
    int n = idx >> 9;               // output col (0..1535)
    int k = idx & 511;
    wqkvT[n * DIMC + k] = (f16)wqkv[k * (3 * DIMC) + n];
  } else {
    int j = idx - NQ;               // 0 .. 512*512-1
    int n = j >> 9;
    int k = j & 511;
    wprojT[n * DIMC + k] = (f16)wproj[k * DIMC + n];
  }
}

// ---------------------------------------------------------------------------
// Kernel 1: fused QKV GEMM + Gaussian-masked softmax attention, one batch
// element per workgroup (8 waves). Wave w owns heads 2w, 2w+1 end-to-end.
// Output: attention result as f16 [B][64][512] into workspace.
// ---------------------------------------------------------------------------
__global__ __launch_bounds__(256, 1)
void gwa_fused_qkv_attn(const float* __restrict__ x,
                        const f16*   __restrict__ wqkvT,
                        const float* __restrict__ bqkv,
                        f16*         __restrict__ aout) {
  __shared__ __align__(16) f16   xs[NTOK * DIMC];            // 64 KB x[b] f16
  __shared__ __align__(16) f16   qs[HEADS * NTOK * HEADDIM]; // 64 KB q [h][m][d]
  __shared__ __align__(16) f16   ks[HEADS * NTOK * HEADDIM]; // 64 KB k [h][n][d] (=B^T)
  __shared__ __align__(16) f16   vts[HEADS * HEADDIM * NTOK];// 64 KB v^T [h][d][n] (=B^T)
  __shared__ __align__(16) float sscr[8][16 * 64];           // 32 KB per-wave S tile
  __shared__ __align__(16) f16   pscr[8][16 * 64];           // 16 KB per-wave P tile
  __shared__ float rsi[NTOK];                                // Gaussian row-sum^-1
  __shared__ float smInv[8][16];                             // softmax 1/sum per row

  const int b    = blockIdx.x;
  const int tid  = threadIdx.x;
  const int w    = tid >> 5;
  const int lane = tid & 31;
  const int nl   = lane & 15;
  const int mh   = (lane >> 4) * 8;

  const float inv2s2 = 1.0f / (2.0f * 0.39f * 0.39f);
  const float scale  = 0.17677669529663687f;  // 32^-0.5

  // --- Gaussian kernel row normalizers (row-sum of unnormalized K) ---
  if (tid < NTOK) {
    float s = 0.f;
    for (int j = 0; j < NTOK; ++j) s += __expf(-dist2(tid, j) * inv2s2);
    rsi[tid] = 1.0f / s;
  }

  // --- Stage x[b] (fp32 -> f16) into LDS ---
  {
    const float4* xp = (const float4*)(x + (size_t)b * NTOK * DIMC);
    v4h* xd = (v4h*)xs;
    for (int i = tid; i < (NTOK * DIMC) / 4; i += 256) {
      float4 t = xp[i];
      v4h h; h[0] = (f16)t.x; h[1] = (f16)t.y; h[2] = (f16)t.z; h[3] = (f16)t.w;
      xd[i] = h;
    }
  }
  __syncthreads();

  // --- Phase B: QKV GEMM, wave-private columns (its own 2 heads x q,k,v) ---
#pragma clang loop unroll(disable)
  for (int part = 0; part < 3; ++part) {       // 0=q 1=k 2=v
#pragma clang loop unroll(disable)
    for (int hh = 0; hh < 2; ++hh) {
      const int h = 2 * w + hh;
#pragma clang loop unroll(disable)
      for (int nt = 0; nt < 2; ++nt) {
        const int ncol = part * DIMC + h * HEADDIM + nt * 16;
        const float bias = bqkv[ncol + nl];
        v8f acc[4];
#pragma unroll
        for (int mt = 0; mt < 4; ++mt)
          for (int i = 0; i < 8; ++i) acc[mt][i] = bias;
#pragma clang loop unroll(disable)
        for (int kt = 0; kt < 16; ++kt) {
          const int kk = kt * 32;
          const v16h bf = load_frag_b(wqkvT + (size_t)ncol * DIMC + kk, lane, DIMC);
#pragma unroll
          for (int mt = 0; mt < 4; ++mt) {
            const v16h af = load_frag_a(xs + (mt * 16) * DIMC + kk, lane, DIMC);
            acc[mt] = wmma_f16(af, bf, acc[mt]);
          }
        }
        // scatter D into attention-friendly layouts (branch hoisted out of
        // the element loops -> one uniform branch per tile, no per-store cmp)
        const int d = nt * 16 + nl;
        if (part == 2) {
          // v^T[h][d][m]: m is contiguous per lane -> packed b128 stores
          f16* dst = vts + (h * HEADDIM + d) * NTOK;
#pragma unroll
          for (int mt = 0; mt < 4; ++mt) {
            v8h pv;
            for (int i = 0; i < 8; ++i) pv[i] = (f16)acc[mt][i];
            *(v8h*)(dst + mt * 16 + mh) = pv;
          }
        } else {
          f16* dst = (part == 0 ? qs : ks) + (h * NTOK) * HEADDIM + d;
#pragma unroll
          for (int mt = 0; mt < 4; ++mt)
            for (int i = 0; i < 8; ++i)
              dst[(mt * 16 + mh + i) * HEADDIM] = (f16)acc[mt][i];
        }
      }
    }
  }

  // --- Phase C: attention for this wave's heads (all data wave-private) ---
#pragma clang loop unroll(disable)
  for (int hh = 0; hh < 2; ++hh) {
    const int h = 2 * w + hh;
    const f16* qh = qs  + h * NTOK * HEADDIM;
    const f16* kh = ks  + h * NTOK * HEADDIM;
    const f16* vh = vts + h * HEADDIM * NTOK;

#pragma clang loop unroll(disable)
    for (int mt = 0; mt < 4; ++mt) {            // 16 query rows at a time
      // S = q @ k^T  (K = 32, single WMMA step per N-tile)
      const v16h af = load_frag_a(qh + (mt * 16) * HEADDIM, lane, HEADDIM);
      v8f s[4];
#pragma unroll
      for (int nt2 = 0; nt2 < 4; ++nt2) {
        v8f z; for (int i = 0; i < 8; ++i) z[i] = 0.f;
        const v16h bf = load_frag_b(kh + (nt2 * 16) * HEADDIM, lane, HEADDIM);
        s[nt2] = wmma_f16(af, bf, z);
      }
      // scale * Gaussian mask (row-normalized), scatter into per-wave scratch
#pragma unroll
      for (int nt2 = 0; nt2 < 4; ++nt2) {
        for (int i = 0; i < 8; ++i) {
          const int m = mt * 16 + mh + i;
          const int n = nt2 * 16 + nl;
          const float g = __expf(-dist2(m, n) * inv2s2) * rsi[m];
          sscr[w][(mh + i) * 64 + n] = s[nt2][i] * scale * g;
        }
      }
      // row softmax: lanes 0..15, one row each (unnormalized exp -> P,
      // the 1/sum is folded into the P@V result afterwards)
      if (lane < 16) {
        const float4* rp = (const float4*)(&sscr[w][lane * 64]);
        float mx = -1e30f;
#pragma clang loop unroll(disable)
        for (int i = 0; i < 16; ++i) {
          float4 t = rp[i];
          mx = fmaxf(mx, fmaxf(fmaxf(t.x, t.y), fmaxf(t.z, t.w)));
        }
        float sum = 0.f;
        v4h* pp = (v4h*)(&pscr[w][lane * 64]);
#pragma clang loop unroll(disable)
        for (int i = 0; i < 16; ++i) {
          float4 t = rp[i];
          float e0 = __expf(t.x - mx), e1 = __expf(t.y - mx);
          float e2 = __expf(t.z - mx), e3 = __expf(t.w - mx);
          sum += (e0 + e1) + (e2 + e3);
          v4h pv; pv[0] = (f16)e0; pv[1] = (f16)e1; pv[2] = (f16)e2; pv[3] = (f16)e3;
          pp[i] = pv;
        }
        smInv[w][lane] = 1.0f / sum;
      }
      // O = P @ V : A = P [16 x 64] (2 K-steps), B^T = v^T [d][token]
      v8f o[2];
#pragma unroll
      for (int nd = 0; nd < 2; ++nd)
        for (int i = 0; i < 8; ++i) o[nd][i] = 0.f;
#pragma unroll
      for (int kc = 0; kc < 64; kc += 32) {
        const v16h pa = load_frag_a(&pscr[w][0] + kc, lane, 64);
        for (int nd = 0; nd < 2; ++nd) {
          const v16h vb = load_frag_b(vh + (nd * 16) * NTOK + kc, lane, NTOK);
          o[nd] = wmma_f16(pa, vb, o[nd]);
        }
      }
      // fold softmax denominator, store f16 to workspace [b][m][h*32+d]
#pragma unroll
      for (int nd = 0; nd < 2; ++nd) {
        for (int i = 0; i < 8; ++i) {
          const int m  = mt * 16 + mh + i;
          const float sc = smInv[w][mh + i];
          aout[((size_t)b * NTOK + m) * DIMC + h * HEADDIM + nd * 16 + nl] =
              (f16)(o[nd][i] * sc);
        }
      }
    }
  }
}

// ---------------------------------------------------------------------------
// Kernel 2: projection GEMM  y = attn_out @ W_proj + b_proj  (fp32 out)
// One batch element per workgroup; wave w covers output cols [64w, 64w+64).
// LDS staging uses the CDNA5 async global->LDS DMA path when available.
// ---------------------------------------------------------------------------
__global__ __launch_bounds__(256, 1)
void gwa_proj(const f16*   __restrict__ aout,
              const f16*   __restrict__ wprojT,
              const float* __restrict__ bproj,
              float*       __restrict__ out) {
  __shared__ __align__(16) f16 as_[NTOK * DIMC];  // 64 KB staged A (f16)

  const int b    = blockIdx.x;
  const int tid  = threadIdx.x;
  const int w    = tid >> 5;
  const int lane = tid & 31;
  const int nl   = lane & 15;
  const int mh   = (lane >> 4) * 8;

#if defined(__gfx1250__) && __has_builtin(__builtin_amdgcn_global_load_async_to_lds_b128)
  {
    typedef __attribute__((address_space(1))) v4i* gptr_v4i;
    typedef __attribute__((address_space(3))) v4i* lptr_v4i;
    const f16* src = aout + (size_t)b * NTOK * DIMC;
#pragma clang loop unroll(disable)
    for (int i = tid; i < (NTOK * DIMC) / 8; i += 256) {
      __builtin_amdgcn_global_load_async_to_lds_b128(
          (gptr_v4i)(__UINTPTR_TYPE__)(src + (size_t)i * 8),
          (lptr_v4i)(as_ + (size_t)i * 8),
          0, 0);
    }
#if __has_builtin(__builtin_amdgcn_s_wait_asynccnt)
    __builtin_amdgcn_s_wait_asynccnt(0);
#else
    asm volatile("s_wait_asynccnt 0" ::: "memory");
#endif
  }
#else
  {
    const uint4* src = (const uint4*)(aout + (size_t)b * NTOK * DIMC);
    uint4* dst = (uint4*)as_;
    for (int i = tid; i < (NTOK * DIMC) / 8; i += 256) dst[i] = src[i];
  }
#endif
  __syncthreads();

#pragma clang loop unroll(disable)
  for (int nt = 0; nt < 4; ++nt) {
    const int ncol = w * 64 + nt * 16;
    const float bias = bproj[ncol + nl];
    v8f acc[4];
#pragma unroll
    for (int mt = 0; mt < 4; ++mt)
      for (int i = 0; i < 8; ++i) acc[mt][i] = bias;
#pragma clang loop unroll(disable)
    for (int kt = 0; kt < 16; ++kt) {
      const int kk = kt * 32;
      const v16h bf = load_frag_b(wprojT + (size_t)ncol * DIMC + kk, lane, DIMC);
#pragma unroll
      for (int mt = 0; mt < 4; ++mt) {
        const v16h af = load_frag_a(as_ + (mt * 16) * DIMC + kk, lane, DIMC);
        acc[mt] = wmma_f16(af, bf, acc[mt]);
      }
    }
#pragma unroll
    for (int mt = 0; mt < 4; ++mt) {
      for (int i = 0; i < 8; ++i) {
        const int m = mt * 16 + mh + i;
        out[(size_t)b * NTOK * DIMC + (size_t)m * DIMC + ncol + nl] = acc[mt][i];
      }
    }
  }
}

// ---------------------------------------------------------------------------
extern "C" void kernel_launch(void* const* d_in, const int* in_sizes, int n_in,
                              void* d_out, int out_size, void* d_ws, size_t ws_size,
                              hipStream_t stream) {
  const float* x     = (const float*)d_in[0];
  const float* wqkv  = (const float*)d_in[1];
  const float* bqkv  = (const float*)d_in[2];
  const float* wproj = (const float*)d_in[3];
  const float* bproj = (const float*)d_in[4];
  float* out = (float*)d_out;

  // workspace layout (f16): wqkvT [1536][512] | wprojT [512][512] | aout [B*64*512]
  f16* wqkvT  = (f16*)d_ws;
  f16* wprojT = wqkvT + (size_t)3 * DIMC * DIMC;
  f16* aout   = wprojT + (size_t)DIMC * DIMC;

  const int prep_elems = 3 * DIMC * DIMC + DIMC * DIMC;  // 1,048,576
  gwa_prep_weights<<<prep_elems / 256, 256, 0, stream>>>(wqkv, wproj, wqkvT, wprojT);
  gwa_fused_qkv_attn<<<BATCH, 256, 0, stream>>>(x, wqkvT, bqkv, aout);
  gwa_proj<<<BATCH, 256, 0, stream>>>(aout, wprojT, bproj, out);
}
